// DenseValueLocalGlobalPartialLM_34368328303141
// MI455X (gfx1250) — compile-verified, matrix operand order checked by reading, not run on hardware
//
#include <hip/hip_runtime.h>
#include <hip/hip_bf16.h>
#include <stdint.h>

// ---- problem dims (fixed by reference) ----
#define B_    2
#define S_    1024
#define V_    32000
#define E_    512
#define H_    1024
#define M_    256
#define P_    4096
#define NTOK  (B_ * S_)
#define LOCW  128
#define CHUNK_ 64
#define NC    16          // S_/CHUNK_

typedef __attribute__((ext_vector_type(16))) __bf16 v16bf;
typedef __attribute__((ext_vector_type(8)))  float  v8f;

union Frag { v16bf v; uint4 q[2]; };

__device__ __forceinline__ unsigned short f2bf(float f) {
    unsigned int u = __float_as_uint(f);
    unsigned int lsb = (u >> 16) & 1u;
    return (unsigned short)((u + 0x7FFFu + lsb) >> 16);
}

// ---- CDNA5 async global->LDS path (guarded: falls back to sync copy) ----
#if defined(__has_builtin)
#  if __has_builtin(__builtin_amdgcn_global_load_async_to_lds_b128) && \
      __has_builtin(__builtin_amdgcn_s_wait_asynccnt)
#    define USE_ASYNC_LDS 1
#  endif
#endif
#ifndef USE_ASYNC_LDS
#  define USE_ASYNC_LDS 0
#endif

#if USE_ASYNC_LDS
// Builtin signature (from hipcc diagnostic): arg0 is int4* in AS1 (global),
// arg1 LDS side, then two immediates (offset, cpol).
typedef int vsi4 __attribute__((vector_size(16)));
typedef __attribute__((address_space(1))) vsi4 vsi4_g;   // global
typedef __attribute__((address_space(3))) vsi4 vsi4_l;   // LDS
#endif

// =====================================================================
// Generic bf16 WMMA GEMM:  C[M,N] = act( A[M,K] @ W[N,K]^T + bias*scale )
// Block tile: 128M x 128N, 8 waves, each wave 16M x 128N (8 accumulators).
// B tile (128 x 32 bf16) staged in LDS (async-to-LDS when available),
// double-buffered, shared by all 8 waves. Dims must satisfy:
//   M % 128 == 0, N % 128 == 0, K % 32 == 0 (true for all call sites).
// ACT: 0 = none, 1 = relu(x)^2.  HAS_ROWSCALE: bias = bias[n]*rowscale[row].
// NT: non-temporal output stores (for the 262MB logits tensor > L2).
// =====================================================================
#define BSTRIDE 40   // shorts per LDS B row (80B: 16B-aligned, conflict-light)

template<int ACT, int HAS_ROWSCALE, int NT>
__global__ __launch_bounds__(256) void gemm_wmma(
    const unsigned short* __restrict__ A, int lda,
    const unsigned short* __restrict__ W, int ldw,
    const float* __restrict__ bias,
    const float* __restrict__ rowscale,
    float* __restrict__ Cf,
    unsigned short* __restrict__ Cb, int ldc,
    int Mdim, int Ndim, int Kdim)
{
    __shared__ __align__(16) unsigned short bt[2][128 * BSTRIDE];  // 20 KB

    const int tid  = threadIdx.x;
    const int lane = tid & 31;
    const int wave = tid >> 5;
    const int m0 = blockIdx.y * 128 + wave * 16;
    const int n0 = blockIdx.x * 128;

    // 16-bit fragment striping: lanes 0-15 rows 0-15 with K=kb..kb+7 and
    // K=16+kb..16+kb+7 (kb = 0 for lanes 0-15, 8 for lanes 16-31).
    const int rsel = lane & 15;
    const int kb   = (lane >> 4) * 8;
    const unsigned short* arow = A + (size_t)(m0 + rsel) * lda + kb;

    // cooperative B stage: 512 16B-chunks; thread moves chunks tid and tid+256.
    // chunk c -> row = c>>2 (0..127), koff = (c&3)*8
    const int r0 = tid >> 2,         kk0 = (tid & 3) * 8;
    const int r1 = (tid + 256) >> 2, kk1 = ((tid + 256) & 3) * 8;
    const unsigned short* g0 = W + (size_t)(n0 + r0) * ldw + kk0;
    const unsigned short* g1 = W + (size_t)(n0 + r1) * ldw + kk1;
    unsigned short* l00 = &bt[0][r0 * BSTRIDE + kk0];
    unsigned short* l01 = &bt[0][r1 * BSTRIDE + kk1];
    unsigned short* l10 = &bt[1][r0 * BSTRIDE + kk0];
    unsigned short* l11 = &bt[1][r1 * BSTRIDE + kk1];

    auto STAGE = [&](int kt) {
        const int k0 = kt << 5;
        unsigned short* d0 = (kt & 1) ? l10 : l00;
        unsigned short* d1 = (kt & 1) ? l11 : l01;
#if USE_ASYNC_LDS
        __builtin_amdgcn_global_load_async_to_lds_b128(
            (vsi4_g*)(void*)(g0 + k0), (vsi4_l*)(void*)d0, 0, 0);
        __builtin_amdgcn_global_load_async_to_lds_b128(
            (vsi4_g*)(void*)(g1 + k0), (vsi4_l*)(void*)d1, 0, 0);
#else
        uint4 t0 = *(const uint4*)(g0 + k0);
        uint4 t1 = *(const uint4*)(g1 + k0);
        *(uint4*)d0 = t0;
        *(uint4*)d1 = t1;
#endif
    };
    auto FENCE = [&]() {
#if USE_ASYNC_LDS
        __builtin_amdgcn_s_wait_asynccnt(0);
#endif
        __syncthreads();
    };

    v8f acc[8] = {v8f{}, v8f{}, v8f{}, v8f{}, v8f{}, v8f{}, v8f{}, v8f{}};
    const int nK = Kdim >> 5;
    STAGE(0);
    FENCE();
    for (int kt = 0; kt < nK; ++kt) {
        const int k0 = kt << 5;
        if (kt + 1 < nK) STAGE(kt + 1);
        Frag a;
        a.q[0] = *(const uint4*)(arow + k0);
        a.q[1] = *(const uint4*)(arow + k0 + 16);
        __builtin_prefetch(arow + k0 + 64, 0, 1);     // global_prefetch_b8
        const unsigned short* bbase = bt[kt & 1];
#pragma unroll
        for (int j = 0; j < 8; ++j) {
            Frag b;
            const unsigned short* bp = bbase + (j * 16 + rsel) * BSTRIDE + kb;
            b.q[0] = *(const uint4*)(bp);
            b.q[1] = *(const uint4*)(bp + 16);
            acc[j] = __builtin_amdgcn_wmma_f32_16x16x32_bf16(
                false, a.v, false, b.v, (short)0, acc[j], false, false);
        }
        FENCE();
    }

    // C/D layout: lane l holds col (l&15); VGPR r holds row r (+8 for lanes>=16)
    const int col   = lane & 15;
    const int rbase = (lane >> 4) * 8;
#pragma unroll
    for (int j = 0; j < 8; ++j) {
        const int n = n0 + j * 16 + col;
        const float bv = bias ? bias[n] : 0.0f;
#pragma unroll
        for (int rr = 0; rr < 8; ++rr) {
            const int row = m0 + rbase + rr;
            const float sc = HAS_ROWSCALE ? rowscale[row] : 1.0f;
            float v = acc[j][rr] + bv * sc;
            if (ACT == 1) { v = v > 0.0f ? v : 0.0f; v = v * v; }
            if (Cf) {
                if (NT) __builtin_nontemporal_store(v, &Cf[(size_t)row * ldc + n]);
                else    Cf[(size_t)row * ldc + n] = v;
            }
            if (Cb) Cb[(size_t)row * ldc + n] = f2bf(v);
        }
    }
}

// =====================================================================
// GRU scan: single workgroup (1024 thr = 32 waves). h kept as bf16 16x1024
// A-tile in LDS (rows 0,1 = batches, rest zero). Each step: gh = h @ whh^T
// via WMMA. K-loop is OUTER with 6 accumulators per wave so each LDS A
// fragment is consumed immediately (no LICM hoist -> no scratch spill).
// =====================================================================
__global__ __launch_bounds__(1024) void gru_scan(
    const float* __restrict__ xp,            // [NTOK, 3H] f32 (x@Wih^T + bih)
    const unsigned short* __restrict__ whh,  // [3H, H] bf16 row-major
    const float* __restrict__ bhh,           // [3H]
    float* __restrict__ states_f,            // [NTOK, H]
    unsigned short* __restrict__ states_b)   // [NTOK, H] bf16
{
    __shared__ __align__(16) unsigned short hb[16][H_];  // bf16 h, zero-padded rows
    __shared__ float gh[B_][3 * H_];
    __shared__ float hf[B_][H_];

    const int tid = threadIdx.x;
    for (int i = tid; i < 16 * H_; i += 1024) ((unsigned short*)hb)[i] = 0;
    for (int i = tid; i < B_ * H_; i += 1024) ((float*)hf)[i] = 0.0f;
    __syncthreads();

    const int lane = tid & 31;
    const int wave = tid >> 5;
    const int rsel = lane & 15;
    const int kb   = (lane >> 4) * 8;
    const unsigned short* hrow  = &hb[rsel][kb];
    const unsigned short* wbase = whh + (size_t)(wave * 96 + rsel) * H_ + kb;

    for (int s = 0; s < S_; ++s) {
        // phase 1: gh = h @ whh^T; wave covers N-range [wave*96, wave*96+96)
        v8f acc[6] = {v8f{}, v8f{}, v8f{}, v8f{}, v8f{}, v8f{}};
        for (int k0 = 0; k0 < H_; k0 += 32) {
            Frag a;
            a.q[0] = *(const uint4*)(hrow + k0);          // ds_load_b128
            a.q[1] = *(const uint4*)(hrow + k0 + 16);
#pragma unroll
            for (int i = 0; i < 6; ++i) {
                Frag b;
                const unsigned short* wp = wbase + (size_t)(i * 16) * H_ + k0;
                b.q[0] = *(const uint4*)(wp);
                b.q[1] = *(const uint4*)(wp + 16);
                acc[i] = __builtin_amdgcn_wmma_f32_16x16x32_bf16(
                    false, a.v, false, b.v, (short)0, acc[i], false, false);
            }
        }
#pragma unroll
        for (int i = 0; i < 6; ++i) {
            const int n0 = (wave * 6 + i) * 16;
            if (lane < 16) {          // row0 -> acc[i][0], row1 -> acc[i][1]
                gh[0][n0 + lane] = acc[i][0];
                gh[1][n0 + lane] = acc[i][1];
            }
        }
        __syncthreads();

        // phase 2: gates (r,z,n order), h update, emit states
        const int j = tid;  // 0..1023
        float hnew[B_];
#pragma unroll
        for (int b = 0; b < B_; ++b) {
            const float* xpt = xp + (size_t)(b * S_ + s) * (3 * H_);
            const float xr = xpt[j], xz = xpt[H_ + j], xn = xpt[2 * H_ + j];
            const float hr = gh[b][j]          + bhh[j];
            const float hz = gh[b][H_ + j]     + bhh[H_ + j];
            const float hn = gh[b][2 * H_ + j] + bhh[2 * H_ + j];
            const float r = 1.0f / (1.0f + __expf(-(xr + hr)));
            const float z = 1.0f / (1.0f + __expf(-(xz + hz)));
            const float n = tanhf(xn + r * hn);
            const float h = (1.0f - z) * n + z * hf[b][j];
            hnew[b] = h;
            states_f[(size_t)(b * S_ + s) * H_ + j] = h;
            states_b[(size_t)(b * S_ + s) * H_ + j] = f2bf(h);
        }
#pragma unroll
        for (int b = 0; b < B_; ++b) { hf[b][j] = hnew[b]; hb[b][j] = f2bf(hnew[b]); }
        __syncthreads();
    }
}

// =====================================================================
// Local windowed attention (<=128 keys per query) — f32 VALU path.
// =====================================================================
__global__ __launch_bounds__(128) void local_attn(
    const float* __restrict__ q, const float* __restrict__ lk,
    const float* __restrict__ lv, float* __restrict__ lctx)
{
    const int t = blockIdx.x, b = t / S_, s = t % S_;
    const int tid = threadIdx.x;
    int kstart = s - LOCW; if (kstart < 0) kstart = 0;
    const int kcount = s - kstart;          // keys [kstart, s)

    __shared__ float sc[LOCW];
    __shared__ float red[128];

    float myscore = -3.4e38f;
    if (tid < kcount) {
        const float* qp = q  + (size_t)t * M_;
        const float* kp = lk + (size_t)(b * S_ + kstart + tid) * M_;
        float d = 0.0f;
        for (int i = 0; i < M_; ++i) d += qp[i] * kp[i];
        myscore = d * (1.0f / 16.0f);       // 1/sqrt(256)
    }
    red[tid] = myscore; __syncthreads();
    for (int st = 64; st > 0; st >>= 1) { if (tid < st) red[tid] = fmaxf(red[tid], red[tid + st]); __syncthreads(); }
    const float mx = red[0]; __syncthreads();
    const float e = (tid < kcount) ? __expf(myscore - mx) : 0.0f;
    red[tid] = e; __syncthreads();
    for (int st = 64; st > 0; st >>= 1) { if (tid < st) red[tid] += red[tid + st]; __syncthreads(); }
    const float ssum = red[0];
    sc[tid] = (kcount > 0 && ssum > 0.0f && tid < kcount) ? e / ssum : 0.0f;
    __syncthreads();

    for (int ee = tid; ee < E_; ee += 128) {
        float a = 0.0f;
        for (int k = 0; k < kcount; ++k)
            a += sc[k] * lv[(size_t)(b * S_ + kstart + k) * E_ + ee];
        lctx[(size_t)t * E_ + ee] = a;
    }
}

// ---- global context helpers ----
__global__ __launch_bounds__(256) void g_summary(
    const float* __restrict__ states_f, float* __restrict__ gsum)
{
    const int bc = blockIdx.x;              // b*NC + c
    const int b = bc / NC, c = bc % NC;
    for (int j = threadIdx.x; j < H_; j += 256) {
        float a = 0.0f;
        for (int r = 0; r < CHUNK_; ++r)
            a += states_f[(size_t)(b * S_ + c * CHUNK_ + r) * H_ + j];
        gsum[(size_t)bc * H_ + j] = a * (1.0f / CHUNK_);
    }
}

__global__ __launch_bounds__(256) void g_kv(
    const float* __restrict__ gsum,
    const float* __restrict__ kw, const float* __restrict__ kbias,
    const float* __restrict__ vw, const float* __restrict__ vbias,
    float* __restrict__ gk, float* __restrict__ gv)
{
    const int bc = blockIdx.x;
    const float* su = gsum + (size_t)bc * H_;
    for (int n = threadIdx.x; n < M_ + E_; n += 256) {
        const float* w = (n < M_) ? (kw + (size_t)n * H_) : (vw + (size_t)(n - M_) * H_);
        float d = 0.0f;
        for (int i = 0; i < H_; ++i) d += su[i] * w[i];
        if (n < M_) gk[(size_t)bc * M_ + n]        = d + kbias[n];
        else        gv[(size_t)bc * E_ + (n - M_)] = d + vbias[n - M_];
    }
}

__global__ __launch_bounds__(128) void g_attn(
    const float* __restrict__ q, const float* __restrict__ gk,
    const float* __restrict__ gv, float* __restrict__ gctx)
{
    const int t = blockIdx.x, b = t / S_, s = t % S_;
    const int tid = threadIdx.x;
    __shared__ float sco[NC], att[NC];
    if (tid < NC) {
        const float* qp = q  + (size_t)t * M_;
        const float* kp = gk + (size_t)(b * NC + tid) * M_;
        float d = 0.0f;
        for (int i = 0; i < M_; ++i) d += qp[i] * kp[i];
        sco[tid] = d * (1.0f / 16.0f);
    }
    __syncthreads();
    if (tid == 0) {
        int nvalid = 0; float mx = -3.4e38f;
        for (int c = 0; c < NC; ++c) {
            const bool valid = (c * CHUNK_ + CHUNK_ - 1) < (s - LOCW);
            if (valid) { ++nvalid; mx = fmaxf(mx, sco[c]); }
        }
        float ssum = 0.0f;
        for (int c = 0; c < NC; ++c) {
            const bool valid = (c * CHUNK_ + CHUNK_ - 1) < (s - LOCW);
            const float ev = valid ? __expf(sco[c] - mx) : 0.0f;
            att[c] = ev; ssum += ev;
        }
        const float inv = (nvalid > 0) ? 1.0f / ssum : 0.0f;
        for (int c = 0; c < NC; ++c) att[c] *= inv;
    }
    __syncthreads();
    for (int e = tid; e < E_; e += 128) {
        float a = 0.0f;
        for (int c = 0; c < NC; ++c)
            a += att[c] * gv[(size_t)(b * NC + c) * E_ + e];
        gctx[(size_t)t * E_ + e] = a;
    }
}

// =====================================================================
// mix softmax + combined feature:
//   comb = base_feat + m0*ls*lctx + m1*gs*gctx  (bf16)
//   rowscale[t] = 1 + m0*ls + m1*gs   (folds the 3x part_b terms into 1 GEMM)
// =====================================================================
__global__ __launch_bounds__(256) void mix_comb(
    const float* __restrict__ states_f,
    const float* __restrict__ mix_w, const float* __restrict__ mix_b,
    const float* __restrict__ base_feat,
    const float* __restrict__ lctx, const float* __restrict__ gctx,
    const float* __restrict__ ls, const float* __restrict__ gs,
    unsigned short* __restrict__ comb_b, float* __restrict__ rowscale)
{
    const int t = blockIdx.x, tid = threadIdx.x;
    __shared__ float r0[256], r1[256];
    __shared__ float m0s, m1s;
    const float* st = states_f + (size_t)t * H_;
    float p0 = 0.0f, p1 = 0.0f;
    for (int j = tid; j < H_; j += 256) {
        const float sv = st[j];
        p0 += sv * mix_w[j];
        p1 += sv * mix_w[H_ + j];
    }
    r0[tid] = p0; r1[tid] = p1; __syncthreads();
    for (int st2 = 128; st2 > 0; st2 >>= 1) {
        if (tid < st2) { r0[tid] += r0[tid + st2]; r1[tid] += r1[tid + st2]; }
        __syncthreads();
    }
    if (tid == 0) {
        const float d0 = r0[0] + mix_b[0], d1 = r1[0] + mix_b[1];
        const float mx = fmaxf(d0, d1);
        const float e0 = __expf(d0 - mx), e1 = __expf(d1 - mx);
        const float m0 = e0 / (e0 + e1);
        m0s = m0 * ls[0];
        m1s = (1.0f - m0) * gs[0];
        rowscale[t] = 1.0f + m0s + m1s;
    }
    __syncthreads();
    const float a0 = m0s, a1 = m1s;
    for (int e = tid; e < E_; e += 256) {
        const float v = base_feat[(size_t)t * E_ + e]
                      + a0 * lctx[(size_t)t * E_ + e]
                      + a1 * gctx[(size_t)t * E_ + e];
        comb_b[(size_t)t * E_ + e] = f2bf(v);
    }
}

// ---- misc small kernels ----
__global__ __launch_bounds__(256) void embed_gather(
    const int* __restrict__ ids, const float* __restrict__ emb,
    unsigned short* __restrict__ xb)
{
    const size_t i = (size_t)blockIdx.x * 256 + threadIdx.x;
    if (i >= (size_t)NTOK * E_) return;
    const int t = (int)(i / E_), e = (int)(i % E_);
    xb[i] = f2bf(emb[(size_t)ids[t] * E_ + e]);
}

__global__ __launch_bounds__(256) void cvt_bf16(
    const float* __restrict__ src, unsigned short* __restrict__ dst, int n)
{
    for (int i = blockIdx.x * 256 + threadIdx.x; i < n; i += gridDim.x * 256)
        dst[i] = f2bf(src[i]);
}

__global__ __launch_bounds__(256) void scatter_add(
    const float* __restrict__ tp, const int* __restrict__ untied,
    float* __restrict__ out)
{
    const size_t idx = (size_t)blockIdx.x * 256 + threadIdx.x;
    if (idx >= (size_t)NTOK * P_) return;
    const int t = (int)(idx / P_), p = (int)(idx % P_);
    const float add = __builtin_nontemporal_load(&tp[idx]);  // one-shot read
    out[(size_t)t * V_ + untied[p]] += add;                  // untied ids unique
}

// =====================================================================
extern "C" void kernel_launch(void* const* d_in, const int* in_sizes, int n_in,
                              void* d_out, int out_size, void* d_ws, size_t ws_size,
                              hipStream_t stream) {
    // inputs (setup_inputs order)
    const int*   input_ids = (const int*)  d_in[0];
    const int*   untied    = (const int*)  d_in[1];
    const float* emb       = (const float*)d_in[2];
    const float* gru_wih   = (const float*)d_in[3];
    const float* gru_whh   = (const float*)d_in[4];
    const float* gru_bih   = (const float*)d_in[5];
    const float* gru_bhh   = (const float*)d_in[6];
    const float* fc_w      = (const float*)d_in[7];
    const float* fc_b      = (const float*)d_in[8];
    const float* proj_w    = (const float*)d_in[9];
    const float* proj_b    = (const float*)d_in[10];
    const float* out_bias  = (const float*)d_in[11];
    const float* part_w    = (const float*)d_in[12];
    const float* part_b    = (const float*)d_in[13];
    const float* q_w       = (const float*)d_in[14];
    const float* q_b       = (const float*)d_in[15];
    const float* k_w       = (const float*)d_in[16];
    const float* k_b       = (const float*)d_in[17];
    const float* v_w       = (const float*)d_in[18];
    const float* v_b       = (const float*)d_in[19];
    const float* mix_w     = (const float*)d_in[20];
    const float* mix_b     = (const float*)d_in[21];
    const float* ls        = (const float*)d_in[22];
    const float* gs        = (const float*)d_in[23];
    float* out = (float*)d_out;

    // workspace carve-up (256B aligned)
    char* ws = (char*)d_ws;
    size_t off = 0;
    auto ALLOC = [&](size_t bytes) -> void* {
        void* p = ws + off;
        off += (bytes + 255) & ~(size_t)255;
        return p;
    };
    unsigned short* wih_b  = (unsigned short*)ALLOC((size_t)3*H_*E_*2);
    unsigned short* whh_b  = (unsigned short*)ALLOC((size_t)3*H_*H_*2);
    unsigned short* fcw_b  = (unsigned short*)ALLOC((size_t)4*E_*H_*2);
    unsigned short* prw_b  = (unsigned short*)ALLOC((size_t)E_*4*E_*2);
    unsigned short* emb_b  = (unsigned short*)ALLOC((size_t)V_*E_*2);
    unsigned short* paw_b  = (unsigned short*)ALLOC((size_t)P_*E_*2);
    unsigned short* qw_b   = (unsigned short*)ALLOC((size_t)M_*H_*2);
    unsigned short* kw_b   = (unsigned short*)ALLOC((size_t)M_*H_*2);
    unsigned short* vw_b   = (unsigned short*)ALLOC((size_t)E_*H_*2);
    unsigned short* x_b    = (unsigned short*)ALLOC((size_t)NTOK*E_*2);
    float*          xp_f   = (float*)ALLOC((size_t)NTOK*3*H_*4);
    float*          st_f   = (float*)ALLOC((size_t)NTOK*H_*4);
    unsigned short* st_b   = (unsigned short*)ALLOC((size_t)NTOK*H_*2);
    unsigned short* head_b = (unsigned short*)ALLOC((size_t)NTOK*4*E_*2);
    float*          bf_f   = (float*)ALLOC((size_t)NTOK*E_*4);
    unsigned short* bf_b   = (unsigned short*)ALLOC((size_t)NTOK*E_*2);
    float*          q_f    = (float*)ALLOC((size_t)NTOK*M_*4);
    float*          lk_f   = (float*)ALLOC((size_t)NTOK*M_*4);
    float*          lv_f   = (float*)ALLOC((size_t)NTOK*E_*4);
    float*          lctx   = (float*)ALLOC((size_t)NTOK*E_*4);
    float*          gctx   = (float*)ALLOC((size_t)NTOK*E_*4);
    float*          gsum   = (float*)ALLOC((size_t)B_*NC*H_*4);
    float*          gk     = (float*)ALLOC((size_t)B_*NC*M_*4);
    float*          gv     = (float*)ALLOC((size_t)B_*NC*E_*4);
    unsigned short* comb_b = (unsigned short*)ALLOC((size_t)NTOK*E_*2);
    float*          rsc    = (float*)ALLOC((size_t)NTOK*4);
    float*          tp     = (float*)ALLOC((size_t)NTOK*P_*4);
    (void)ws_size; (void)in_sizes; (void)n_in; (void)out_size;

    auto CVT = [&](const float* s, unsigned short* d, int n) {
        cvt_bf16<<<(n + 255) / 256, 256, 0, stream>>>(s, d, n);
    };
    // weight down-conversions (f32 -> bf16)
    CVT(gru_wih, wih_b, 3*H_*E_);
    CVT(gru_whh, whh_b, 3*H_*H_);
    CVT(fc_w,    fcw_b, 4*E_*H_);
    CVT(proj_w,  prw_b, E_*4*E_);
    CVT(emb,     emb_b, V_*E_);
    CVT(part_w,  paw_b, P_*E_);
    CVT(q_w,     qw_b,  M_*H_);
    CVT(k_w,     kw_b,  M_*H_);
    CVT(v_w,     vw_b,  E_*H_);

    // x = emb[ids] (bf16)
    embed_gather<<<(NTOK*E_ + 255)/256, 256, 0, stream>>>(input_ids, emb, x_b);

    // xp = x @ Wih^T + bih   [2048,3072]
    gemm_wmma<0,0,0><<<dim3(3*H_/128, NTOK/128), 256, 0, stream>>>(
        x_b, E_, wih_b, E_, gru_bih, nullptr, xp_f, nullptr, 3*H_, NTOK, 3*H_, E_);

    // GRU recurrence (sequential over S)
    gru_scan<<<1, 1024, 0, stream>>>(xp_f, whh_b, gru_bhh, st_f, st_b);

    // head = relu(states @ fc^T + b)^2  -> bf16  [2048,2048]
    gemm_wmma<1,0,0><<<dim3(4*E_/128, NTOK/128), 256, 0, stream>>>(
        st_b, H_, fcw_b, H_, fc_b, nullptr, nullptr, head_b, 4*E_, NTOK, 4*E_, H_);

    // base_feat = head @ proj^T + b  [2048,512] (f32 + bf16)
    gemm_wmma<0,0,0><<<dim3(E_/128, NTOK/128), 256, 0, stream>>>(
        head_b, 4*E_, prw_b, 4*E_, proj_b, nullptr, bf_f, bf_b, E_, NTOK, E_, 4*E_);

    // q / lk / lv projections
    gemm_wmma<0,0,0><<<dim3(M_/128, NTOK/128), 256, 0, stream>>>(
        st_b, H_, qw_b, H_, q_b, nullptr, q_f, nullptr, M_, NTOK, M_, H_);
    gemm_wmma<0,0,0><<<dim3(M_/128, NTOK/128), 256, 0, stream>>>(
        st_b, H_, kw_b, H_, k_b, nullptr, lk_f, nullptr, M_, NTOK, M_, H_);
    gemm_wmma<0,0,0><<<dim3(E_/128, NTOK/128), 256, 0, stream>>>(
        st_b, H_, vw_b, H_, v_b, nullptr, lv_f, nullptr, E_, NTOK, E_, H_);

    // local + global attention
    local_attn<<<NTOK, 128, 0, stream>>>(q_f, lk_f, lv_f, lctx);
    g_summary<<<B_*NC, 256, 0, stream>>>(st_f, gsum);
    g_kv<<<B_*NC, 256, 0, stream>>>(gsum, k_w, k_b, v_w, v_b, gk, gv);
    g_attn<<<NTOK, 128, 0, stream>>>(q_f, gk, gv, gctx);

    // mix softmax + combined feature + bias row-scale
    mix_comb<<<NTOK, 256, 0, stream>>>(st_f, mix_w, mix_b, bf_f, lctx, gctx,
                                       ls, gs, comb_b, rsc);

    // total_partial = comb @ part^T + part_b * rowscale  [2048,4096]
    gemm_wmma<0,1,0><<<dim3(P_/128, NTOK/128), 256, 0, stream>>>(
        comb_b, E_, paw_b, E_, part_b, rsc, tp, nullptr, P_, NTOK, P_, E_);

    // base_logits = base_feat @ emb^T + out_bias -> d_out (NT stores, > L2)
    gemm_wmma<0,0,1><<<dim3(V_/128, NTOK/128), 256, 0, stream>>>(
        bf_b, E_, emb_b, E_, out_bias, nullptr, out, nullptr, V_, NTOK, V_, E_);

    // scatter-add partial logits into untied columns
    scatter_add<<<(NTOK*P_ + 255)/256, 256, 0, stream>>>(tp, untied, out);
}
